// STiSAN_12773232738829
// MI455X (gfx1250) — matrix operand order repaired and checked by reading, not compiled, not run on hardware
//
#include <hip/hip_runtime.h>

// CDNA5 / gfx1250 fused causal multi-head attention.
// All four matmul stages (Wq/Wk/Wv proj, Q.K^T, P.V, Wo proj) go through
// v_wmma_f32_16x16x32_f16 (wave32). All WMMA fragments are gathered from LDS
// with CONTIGUOUS 16B chunks (transposed weight/V staging) so they lower to
// ds_load_b128. Intermediates use an [H][B*S][D] layout so every global
// store/load offset is linear (no div/mod-by-100 address math in the hot
// loops). Softmax is fp32 VALU with 16-lane butterfly shuffles. The causal
// mask input is never read (mask == tril, computed analytically).

typedef __attribute__((ext_vector_type(16))) _Float16 v16h;
typedef __attribute__((ext_vector_type(8)))  float    v8f;

constexpr int Bsz  = 64;
constexpr int Ssz  = 100;
constexpr int Fsz  = 128;
constexpr int Hn   = 8;
constexpr int Dh   = 16;           // == WMMA tile dim
constexpr int Spad = 112;          // 7 tiles of 16
constexpr int Mrows = Bsz * Ssz;   // 6400 global rows
constexpr float ATT_SCALE = 0.25f; // 16^-0.5

constexpr int LDA = 136;  // padded ld (halves) for A-side LDS tiles (272B rows)
constexpr int LDK = 136;  // padded ld (halves) for transposed B-side tiles

// ---- WMMA fragment builders (contiguous k = 16h + s order in A and B, wave32) -------------------

// A fragment: 16x32 f16 tile at (row0, k0) of row-major [*, ld].
// lane 0-15: M=lane, K 0..15 ; lane 16-31: M=lane-16, K 16..31.
// Two contiguous 16B chunks per lane -> 2x ds_load_b128.
__device__ __forceinline__ v16h make_a_frag(const _Float16* s, int ld, int row0, int k0) {
    const int lane = threadIdx.x & 31;
    const int m    = row0 + (lane & 15);
    const int kb   = (lane & 16) ? 16 : 0;
    const _Float16* p0 = s + m * ld + k0 + kb;        // K = kb .. kb+7
    const _Float16* p1 = s + m * ld + k0 + kb + 8;    // K = kb+8 .. kb+15
    v16h a;
#pragma unroll
    for (int e = 0; e < 8; ++e) a[e] = p0[e];
#pragma unroll
    for (int e = 0; e < 8; ++e) a[8 + e] = p1[e];
    return a;
}

// A fragment with contraction dim 16 zero-padded to 32 (Q, D=16):
// lanes 0-15 load K 0..15 (one contiguous 32B run); lanes 16-31 are the zero pad.
__device__ __forceinline__ v16h make_aQ_frag(const _Float16* s, int row0) {
    const int lane = threadIdx.x & 31;
    const int m    = row0 + (lane & 15);
    const _Float16* p = s + m * Dh;
    v16h a;
#pragma unroll
    for (int e = 0; e < 16; ++e) a[e] = (lane & 16) ? (_Float16)0.f : p[e];
    return a;
}

// B fragment from a TRANSPOSED source: sT[n][k] == B[k][n], row-major ld=ldk.
// lane 0-15: N=lane, K-group 0 ; lane 16-31: N=lane-16, K-group 16.
// 16 contiguous halves per lane -> 2x ds_load_b128.
__device__ __forceinline__ v16h make_bT_frag(const _Float16* sT, int ldk, int k0, int n0) {
    const int lane = threadIdx.x & 31;
    const int n    = n0 + (lane & 15);
    const int kg   = (lane & 16) ? 16 : 0;
    const _Float16* p = sT + n * ldk + k0 + kg;
    v16h b;
#pragma unroll
    for (int e = 0; e < 16; ++e) b[e] = p[e];
    return b;
}

// B fragment for K^T with K-dim 16 zero-padded to 32. Source s[key][Dh]
// is already B-transposed (B[k][n] = s[n][k]). Lanes 0-15 (K-group 0) load
// one contiguous 32B run; lanes 16-31 (K-group 16..31) are the zero pad.
__device__ __forceinline__ v16h make_bKT_frag(const _Float16* s, int n0) {
    const int lane = threadIdx.x & 31;
    const int n    = n0 + (lane & 15);
    v16h b;
    if ((lane & 16) == 0) {
        const _Float16* p = s + n * Dh;
#pragma unroll
        for (int e = 0; e < 16; ++e) b[e] = p[e];
    } else {
#pragma unroll
        for (int e = 0; e < 16; ++e) b[e] = (_Float16)0.f;
    }
    return b;
}

// ---- Kernel 1: fused QKV projection ---------------------------------------
// grid (400, 3); block 256 = 8 waves. Each block: one 16-row M tile of X,
// each wave: one head (16-col N tile), K=128 as 4 chained K32 WMMAs.
// Output layout [H][B*S][D]: store offsets are LINEAR in the global row.
__global__ void __launch_bounds__(256)
qkv_proj_kernel(const float* __restrict__ q, const float* __restrict__ k,
                const float* __restrict__ v,
                const float* __restrict__ Wq, const float* __restrict__ Wk,
                const float* __restrict__ Wv,
                const float* __restrict__ bq, const float* __restrict__ bk,
                const float* __restrict__ bv,
                _Float16* __restrict__ qh, _Float16* __restrict__ kh,
                _Float16* __restrict__ vh) {
    __shared__ _Float16 Xs[16 * LDA];        // A tile, padded rows
    __shared__ _Float16 Wt[Fsz * LDK];       // transposed weights: Wt[n][k]

    const int which = blockIdx.y;
    const float* X    = (which == 0) ? q  : (which == 1) ? k  : v;
    const float* W    = (which == 0) ? Wq : (which == 1) ? Wk : Wv;
    const float* bias = (which == 0) ? bq : (which == 1) ? bk : bv;
    _Float16*    out  = (which == 0) ? qh : (which == 1) ? kh : vh;

    const int m0  = blockIdx.x * 16;
    const int tid = threadIdx.x;

    __builtin_prefetch(W + tid * 16, 0, 0);  // global_prefetch_b8

    for (int i = tid; i < 16 * Fsz; i += 256)
        Xs[(i / Fsz) * LDA + (i % Fsz)] =
            (_Float16)X[(size_t)(m0 + i / Fsz) * Fsz + (i % Fsz)];
    for (int i = tid; i < Fsz * Fsz; i += 256) {
        const int kk = i / Fsz, nn = i % Fsz;
        Wt[nn * LDK + kk] = (_Float16)W[i];  // transpose on stage
    }
    __syncthreads();

    const int wv   = tid >> 5;   // wave id == head id == N tile
    const int lane = tid & 31;
    const int n0   = wv * 16;

    v8f acc = {};
#pragma unroll
    for (int kc = 0; kc < 4; ++kc) {
        v16h a = make_a_frag(Xs, LDA, 0, kc * 32);
        v16h b = make_bT_frag(Wt, LDK, kc * 32, n0);
        acc = __builtin_amdgcn_wmma_f32_16x16x32_f16(false, a, false, b,
                                                     (short)0, acc, false, false);
    }

    const int   N     = lane & 15;
    const int   Mbase = (lane & 16) ? 8 : 0;
    const float bb    = bias[n0 + N];
    // [H][B*S][D]: offset linear in row m = m0 + Mbase + r
    const size_t base = ((size_t)wv * Mrows + m0 + Mbase) * Dh + N;
#pragma unroll
    for (int r = 0; r < 8; ++r)
        out[base + (size_t)r * Dh] = (_Float16)(acc[r] + bb);
}

// ---- Kernel 2: per-(b,h) causal attention ---------------------------------
// grid (B*H); block 224 = 7 waves, one query tile (16 rows) per wave.
__global__ void __launch_bounds__(224)
attn_kernel(const _Float16* __restrict__ qh, const _Float16* __restrict__ kh,
            const _Float16* __restrict__ vh, _Float16* __restrict__ ctx) {
    __shared__ _Float16 Qs[Spad * Dh];       // 112x16 (query rows)
    __shared__ _Float16 Ks[Spad * Dh];       // 112x16 (key rows == B^T layout)
    __shared__ _Float16 Vt[Dh * LDK];        // transposed V: Vt[d][key], 16x136
    __shared__ _Float16 Ps[7][16 * LDA];     // per-wave probs, padded rows

    const int bh   = blockIdx.x;             // b*H + h
    const int bidx = bh >> 3;                // Hn == 8
    const int h    = bh & 7;
    const int tid  = threadIdx.x;
    // [H][B*S][D]: per-(b,h) block is still 100 contiguous rows
    const size_t src_off = ((size_t)h * Mrows + (size_t)bidx * Ssz) * Dh;
    const _Float16* qsrc = qh + src_off;
    const _Float16* ksrc = kh + src_off;
    const _Float16* vsrc = vh + src_off;

    for (int i = tid; i < Spad * Dh; i += 224) {
        const int s = i / Dh;
        Qs[i] = (s < Ssz) ? qsrc[i] : (_Float16)0.f;
        Ks[i] = (s < Ssz) ? ksrc[i] : (_Float16)0.f;
    }
    for (int i = tid; i < Dh * 128; i += 224) {
        const int d = i / 128, s = i % 128;  // transpose V on stage
        Vt[d * LDK + s] = (s < Ssz) ? vsrc[s * Dh + d] : (_Float16)0.f;
    }
    __syncthreads();

    const int w     = tid >> 5;              // wave id == query tile
    const int lane  = tid & 31;
    const int qt    = w;
    const int q0    = qt * 16;
    const int N     = lane & 15;
    const int Mbase = (lane & 16) ? 8 : 0;

    // ---- scores = Q . K^T (wave-uniform branches: EXEC all-1s at WMMA) ----
    float sc[7][8];
    const v16h aQ = make_aQ_frag(Qs, q0);
#pragma unroll
    for (int kt = 0; kt < 7; ++kt) {
        if (kt <= qt) {
            v16h bK = make_bKT_frag(Ks, kt * 16);
            v8f  c  = {};
            c = __builtin_amdgcn_wmma_f32_16x16x32_f16(false, aQ, false, bK,
                                                       (short)0, c, false, false);
#pragma unroll
            for (int r = 0; r < 8; ++r) sc[kt][r] = c[r];
        } else {
#pragma unroll
            for (int r = 0; r < 8; ++r) sc[kt][r] = -1e9f;
        }
    }

    // ---- scale + analytic causal mask (mask tensor never read) ----
#pragma unroll
    for (int kt = 0; kt < 7; ++kt)
#pragma unroll
        for (int r = 0; r < 8; ++r) {
            const int qi = q0 + Mbase + r;
            const int ki = kt * 16 + N;
            sc[kt][r] = (ki <= qi && qi < Ssz && ki < Ssz) ? sc[kt][r] * ATT_SCALE
                                                           : -1e9f;
        }

    // ---- row softmax: 16-lane butterflies (xor<16 stays in N-group) ----
    float mx[8], sm[8];
#pragma unroll
    for (int r = 0; r < 8; ++r) mx[r] = -3.4e38f;
#pragma unroll
    for (int kt = 0; kt < 7; ++kt)
#pragma unroll
        for (int r = 0; r < 8; ++r) mx[r] = fmaxf(mx[r], sc[kt][r]);
#pragma unroll
    for (int off = 1; off < 16; off <<= 1)
#pragma unroll
        for (int r = 0; r < 8; ++r)
            mx[r] = fmaxf(mx[r], __shfl_xor(mx[r], off, 32));

#pragma unroll
    for (int r = 0; r < 8; ++r) sm[r] = 0.f;
#pragma unroll
    for (int kt = 0; kt < 7; ++kt)
#pragma unroll
        for (int r = 0; r < 8; ++r) {
            const float p = __expf(sc[kt][r] - mx[r]);
            sc[kt][r] = p;
            sm[r] += p;
        }
#pragma unroll
    for (int off = 1; off < 16; off <<= 1)
#pragma unroll
        for (int r = 0; r < 8; ++r) sm[r] += __shfl_xor(sm[r], off, 32);

    float inv[8];
#pragma unroll
    for (int r = 0; r < 8; ++r) inv[r] = 1.f / sm[r];

    _Float16* Pw = &Ps[w][0];
#pragma unroll
    for (int kt = 0; kt < 7; ++kt)
#pragma unroll
        for (int r = 0; r < 8; ++r)
            Pw[(Mbase + r) * LDA + kt * 16 + N] = (_Float16)(sc[kt][r] * inv[r]);
#pragma unroll
    for (int r = 0; r < 8; ++r)              // zero key pad 112..127
        Pw[(Mbase + r) * LDA + 112 + N] = (_Float16)0.f;

    // ---- context = P . V : 4 chained K32 WMMAs over 128 padded keys ----
    v8f o = {};
#pragma unroll
    for (int ch = 0; ch < 4; ++ch) {
        v16h aP = make_a_frag(Pw, LDA, 0, ch * 32);
        v16h bV = make_bT_frag(Vt, LDK, ch * 32, 0);
        o = __builtin_amdgcn_wmma_f32_16x16x32_f16(false, aP, false, bV,
                                                   (short)0, o, false, false);
    }

    // ctx is [B,S,F] f16 (heads re-interleaved); base + r*Fsz, no div/mod.
    const size_t cbase = ((size_t)bidx * Ssz + q0 + Mbase) * Fsz + h * Dh + N;
#pragma unroll
    for (int r = 0; r < 8; ++r) {
        const int s = q0 + Mbase + r;
        if (s < Ssz)
            ctx[cbase + (size_t)r * Fsz] = (_Float16)o[r];
    }
}

// ---- Kernel 3: output projection out = ctx @ Wo + bo (fp32 out) -----------
__global__ void __launch_bounds__(256)
out_proj_kernel(const _Float16* __restrict__ ctx, const float* __restrict__ Wo,
                const float* __restrict__ bo, float* __restrict__ out) {
    __shared__ _Float16 Cs[16 * LDA];
    __shared__ _Float16 Wt[Fsz * LDK];       // transposed Wo

    const int m0  = blockIdx.x * 16;
    const int tid = threadIdx.x;

    __builtin_prefetch(Wo + tid * 16, 0, 0);

    for (int i = tid; i < 16 * Fsz; i += 256)
        Cs[(i / Fsz) * LDA + (i % Fsz)] = ctx[(size_t)m0 * Fsz + i];
    for (int i = tid; i < Fsz * Fsz; i += 256) {
        const int kk = i / Fsz, nn = i % Fsz;
        Wt[nn * LDK + kk] = (_Float16)Wo[i];
    }
    __syncthreads();

    const int wv = tid >> 5, lane = tid & 31, n0 = wv * 16;

    v8f acc = {};
#pragma unroll
    for (int kc = 0; kc < 4; ++kc) {
        v16h a = make_a_frag(Cs, LDA, 0, kc * 32);
        v16h b = make_bT_frag(Wt, LDK, kc * 32, n0);
        acc = __builtin_amdgcn_wmma_f32_16x16x32_f16(false, a, false, b,
                                                     (short)0, acc, false, false);
    }

    const int    N     = lane & 15;
    const int    Mbase = (lane & 16) ? 8 : 0;
    const float  bb    = bo[n0 + N];
    const size_t base  = (size_t)(m0 + Mbase) * Fsz + n0 + N;
#pragma unroll
    for (int r = 0; r < 8; ++r)
        out[base + (size_t)r * Fsz] = acc[r] + bb;
}

// ---- host launcher --------------------------------------------------------
extern "C" void kernel_launch(void* const* d_in, const int* in_sizes, int n_in,
                              void* d_out, int out_size, void* d_ws, size_t ws_size,
                              hipStream_t stream) {
    (void)in_sizes; (void)n_in; (void)out_size; (void)ws_size;
    const float* q  = (const float*)d_in[0];
    const float* k  = (const float*)d_in[1];
    const float* v  = (const float*)d_in[2];
    // d_in[3] == mask: unused (causal tril computed analytically in-kernel)
    const float* Wq = (const float*)d_in[4];
    const float* bq = (const float*)d_in[5];
    const float* Wk = (const float*)d_in[6];
    const float* bk = (const float*)d_in[7];
    const float* Wv = (const float*)d_in[8];
    const float* bv = (const float*)d_in[9];
    const float* Wo = (const float*)d_in[10];
    const float* bo = (const float*)d_in[11];

    const size_t per_bytes = (size_t)Hn * Mrows * Dh * sizeof(_Float16);
    char* ws = (char*)d_ws;
    _Float16* qh  = (_Float16*)(ws);
    _Float16* kh  = (_Float16*)(ws + per_bytes);
    _Float16* vh  = (_Float16*)(ws + 2 * per_bytes);
    _Float16* ctx = (_Float16*)(ws + 3 * per_bytes);

    // 6400 rows / 16 = 400 M-tiles
    hipLaunchKernelGGL(qkv_proj_kernel, dim3(400, 3), dim3(256), 0, stream,
                       q, k, v, Wq, Wk, Wv, bq, bk, bv, qh, kh, vh);
    hipLaunchKernelGGL(attn_kernel, dim3(Bsz * Hn), dim3(224), 0, stream,
                       qh, kh, vh, ctx);
    hipLaunchKernelGGL(out_proj_kernel, dim3(400), dim3(256), 0, stream,
                       ctx, Wo, bo, (float*)d_out);
}
